// Encoder_9818295238798
// MI455X (gfx1250) — matrix operand order repaired and checked by reading, not compiled, not run on hardware
//
#include <hip/hip_runtime.h>
#include <hip/hip_bf16.h>

#define B_ 32
#define S_ 2048
#define D_ 64
#define H_ 256
#define SCALE_ 0.125f

typedef __attribute__((ext_vector_type(16))) _Float16 v16h;
typedef __attribute__((ext_vector_type(8)))  _Float16 v8h;
typedef __attribute__((ext_vector_type(8)))  float    v8f;
typedef __attribute__((ext_vector_type(4)))  float    v4f;

union HV { v16h v; v8h h[2]; };

static __device__ __forceinline__ v8f vzero8() {
  v8f z;
#pragma unroll
  for (int j = 0; j < 8; ++j) z[j] = 0.0f;
  return z;
}

static __device__ __forceinline__ v8f wmma_f16(v16h a, v16h b, v8f c) {
  // D = A(16x32 f16) * B(32x16 f16) + C(16x16 f32)
  return __builtin_amdgcn_wmma_f32_16x16x32_f16(
      /*neg_a=*/false, a, /*neg_b=*/false, b,
      /*c_mod=*/(short)0, c, /*reuse_a=*/false, /*reuse_b=*/false);
}

// CDNA5 async global->LDS copy (tracked by ASYNCcnt). GV addressing mode:
//   LDS address taken from a VGPR; global addr from VGPR pair.
// Truncating a flat shared pointer to 32 bits yields the LDS byte offset
// (LDS aperture keeps the offset in addr[31:0], ISA 10.2).
static __device__ __forceinline__ void async_copy_b128(void* lds_dst,
                                                       const void* gsrc) {
  unsigned l = (unsigned)(uintptr_t)lds_dst;
  unsigned long long g = (unsigned long long)(uintptr_t)gsrc;
  asm volatile("global_load_async_to_lds_b128 %0, %1, off"
               :: "v"(l), "v"(g)
               : "memory");
}

static __device__ __forceinline__ void wait_async0() {
  asm volatile("s_wait_asynccnt 0x0" ::: "memory");
}

static __device__ __forceinline__ void wait_ds0() {
  asm volatile("s_wait_dscnt 0x0" ::: "memory");
}

// ---------------------------------------------------------------------------
// Kernel 1: embedding gather + Q/K/V projections (WMMA), writes f16
//   Qg, Kg : [b][s][d] row-major f16
//   Vg     : [b][d][t] (transposed) f16  -> contiguous B-fragment reads later
// ---------------------------------------------------------------------------
__global__ __launch_bounds__(256) void enc_qkv_kernel(
    const int* __restrict__ x, const float* __restrict__ emb,
    const float* __restrict__ wq, const float* __restrict__ bq,
    const float* __restrict__ wk, const float* __restrict__ bk,
    const float* __restrict__ wv, const float* __restrict__ bv,
    _Float16* __restrict__ Qg, _Float16* __restrict__ Kg,
    _Float16* __restrict__ Vg) {
  __shared__ __align__(32) _Float16 WT[3][64 * 64];  // [n][d] transposed f16

  const int tid  = threadIdx.x;
  const int lane = tid & 31;
  const int wave = tid >> 5;
  const int ln   = lane & 15;   // A-frag row / C-frag col
  const int hi   = lane >> 4;

  // stage transposed f16 weights
  const float* Ws[3] = {wq, wk, wv};
#pragma unroll
  for (int m = 0; m < 3; ++m)
    for (int idx = tid; idx < 64 * 64; idx += 256) {
      int d = idx >> 6, n = idx & 63;
      WT[m][n * 64 + d] = (_Float16)Ws[m][idx];
    }
  __syncthreads();

  const int tokBase = blockIdx.x * 128 + wave * 16;  // flat token id
  const int b  = tokBase / S_;
  const int s0 = tokBase % S_;

  // A-fragments of the embedding tile (16 tokens x 64 feats, two K=32 chunks)
  const int tok = x[tokBase + ln];
  const float* erow = emb + (size_t)tok * D_;
  HV a[2];
#pragma unroll
  for (int c = 0; c < 2; ++c) {
    int cb = 32 * c + hi * 16;
    v4f f0a = *(const v4f*)(erow + cb);
    v4f f0b = *(const v4f*)(erow + cb + 4);
    v4f f1a = *(const v4f*)(erow + cb + 8);
    v4f f1b = *(const v4f*)(erow + cb + 12);
#pragma unroll
    for (int j = 0; j < 4; ++j) {
      a[c].h[0][j]     = (_Float16)f0a[j];
      a[c].h[0][j + 4] = (_Float16)f0b[j];
      a[c].h[1][j]     = (_Float16)f1a[j];
      a[c].h[1][j + 4] = (_Float16)f1b[j];
    }
  }

#pragma unroll
  for (int mtx = 0; mtx < 3; ++mtx) {
    const float* bias = (mtx == 0) ? bq : (mtx == 1) ? bk : bv;
#pragma unroll
    for (int nt = 0; nt < 4; ++nt) {
      v8f acc = vzero8();
#pragma unroll
      for (int c = 0; c < 2; ++c) {
        v16h bf = *(const v16h*)(&WT[mtx][(nt * 16 + ln) * 64 + c * 32 + hi * 16]);
        acc = wmma_f16(a[c].v, bf, acc);
      }
      const float bval = bias[nt * 16 + ln];
      if (mtx < 2) {
        _Float16* og = (mtx == 0) ? Qg : Kg;
#pragma unroll
        for (int i = 0; i < 8; ++i) {
          int r = i + 8 * hi;
          og[(size_t)(tokBase + r) * D_ + nt * 16 + ln] = (_Float16)(acc[i] + bval);
        }
      } else {
#pragma unroll
        for (int i = 0; i < 8; ++i) {
          int r = i + 8 * hi;
          Vg[((size_t)b * D_ + nt * 16 + ln) * S_ + (s0 + r)] = (_Float16)(acc[i] + bval);
        }
      }
    }
  }
}

// ---------------------------------------------------------------------------
// Kernel 2: flash attention. Block = (b, 128 queries), wave = 16-query tile.
// K/V chunks are double-buffered via CDNA5 async global->LDS copies
// (ASYNCcnt) so the next 32-key block streams in while WMMAs run.
// Writes upd as f16 [b][s][d].
// ---------------------------------------------------------------------------
__global__ __launch_bounds__(256) void enc_attn_kernel(
    const _Float16* __restrict__ Qg, const _Float16* __restrict__ Kg,
    const _Float16* __restrict__ Vg, _Float16* __restrict__ Ug) {
  __shared__ __align__(32) _Float16 Kt[2][32 * 64];   // [t][d] chunks
  __shared__ __align__(32) _Float16 Vt[2][64 * 32];   // [d][t] chunks
  __shared__ __align__(32) _Float16 Pb[8][16 * 32];   // per-wave C->A bounce

  const int tid  = threadIdx.x;
  const int lane = tid & 31;
  const int wave = tid >> 5;
  const int ln   = lane & 15;
  const int hi   = lane >> 4;

  const int tilesPerB = S_ / 128;
  const int b  = blockIdx.x / tilesPerB;
  const int q0 = (blockIdx.x % tilesPerB) * 128 + wave * 16;

  const _Float16* Kb = Kg + (size_t)b * S_ * D_;
  const _Float16* Vb = Vg + (size_t)b * D_ * S_;

  // per-thread source/dest slots for async staging (16B each, 8KB per chunk)
  const int vd = tid >> 2, vtoff = (tid & 3) * 8;

  // kick off chunk 0
  async_copy_b128((v8h*)Kt[0] + tid, (const v8h*)Kb + tid);
  async_copy_b128(Vt[0] + vd * 32 + vtoff, Vb + (size_t)vd * S_ + vtoff);

  // Q A-fragments (rows = queries, K-dim = d)
  const _Float16* qrow = Qg + ((size_t)b * S_ + q0 + ln) * D_;
  HV qa[2];
#pragma unroll
  for (int c = 0; c < 2; ++c) {
    int cb = 32 * c + hi * 16;
    qa[c].h[0] = *(const v8h*)(qrow + cb);
    qa[c].h[1] = *(const v8h*)(qrow + cb + 8);
  }

  v8f acc[4];
#pragma unroll
  for (int n = 0; n < 4; ++n) acc[n] = vzero8();
  float m[8], l[8];
#pragma unroll
  for (int i = 0; i < 8; ++i) { m[i] = -1e30f; l[i] = 0.0f; }

  _Float16* pb = &Pb[wave][0];

#pragma unroll 1
  for (int c0 = 0; c0 < S_; c0 += 32) {
    const int cur = (c0 >> 5) & 1;
    // own async copies done -> barrier makes every thread's copies visible
    wait_async0();
    __syncthreads();
    // stream in the next chunk into the other buffer while we compute
    if (c0 + 32 < S_) {
      const int nxt = cur ^ 1;
      async_copy_b128((v8h*)Kt[nxt] + tid,
                      (const v8h*)(Kb + (size_t)(c0 + 32) * D_) + tid);
      async_copy_b128(Vt[nxt] + vd * 32 + vtoff,
                      Vb + (size_t)vd * S_ + (c0 + 32) + vtoff);
    }
    const _Float16* kt = Kt[cur];
    const _Float16* vt = Vt[cur];

    // ---- scores: s = Q * K^T (two 16-key tiles) ----
    v8f s0 = vzero8(), s1 = vzero8();
#pragma unroll
    for (int kc = 0; kc < 2; ++kc) {
      v16h bf0 = *(const v16h*)(kt + (0 * 16 + ln) * 64 + kc * 32 + hi * 16);
      v16h bf1 = *(const v16h*)(kt + (1 * 16 + ln) * 64 + kc * 32 + hi * 16);
      s0 = wmma_f16(qa[kc].v, bf0, s0);
      s1 = wmma_f16(qa[kc].v, bf1, s1);
    }
#pragma unroll
    for (int i = 0; i < 8; ++i) { s0[i] *= SCALE_; s1[i] *= SCALE_; }

    // ---- online softmax per row (rows live across 16-lane halves) ----
    float fac[8];
#pragma unroll
    for (int i = 0; i < 8; ++i) {
      float cm = fmaxf(s0[i], s1[i]);
#pragma unroll
      for (int off = 8; off >= 1; off >>= 1) cm = fmaxf(cm, __shfl_xor(cm, off, 16));
      float mn = fmaxf(m[i], cm);
      fac[i] = __expf(m[i] - mn);
      float p0 = __expf(s0[i] - mn);
      float p1 = __expf(s1[i] - mn);
      s0[i] = p0; s1[i] = p1;
      float rs = p0 + p1;
#pragma unroll
      for (int off = 8; off >= 1; off >>= 1) rs += __shfl_xor(rs, off, 16);
      l[i] = l[i] * fac[i] + rs;
      m[i] = mn;
    }
#pragma unroll
    for (int n = 0; n < 4; ++n)
#pragma unroll
      for (int i = 0; i < 8; ++i) acc[n][i] *= fac[i];

    // ---- bounce P through LDS: C layout -> A-fragment layout ----
#pragma unroll
    for (int i = 0; i < 8; ++i) {
      int r = i + 8 * hi;
      pb[r * 32 + ln]      = (_Float16)s0[i];
      pb[r * 32 + ln + 16] = (_Float16)s1[i];
    }
    wait_ds0();
    HV pa;
    pa.h[0] = *(const v8h*)(pb + ln * 32 + hi * 16);
    pa.h[1] = *(const v8h*)(pb + ln * 32 + hi * 16 + 8);

    // ---- O += P * V ----
#pragma unroll
    for (int n = 0; n < 4; ++n) {
      v16h vf = *(const v16h*)(vt + (n * 16 + ln) * 32 + hi * 16);
      acc[n] = wmma_f16(pa.v, vf, acc[n]);
    }
  }

  // normalize and store upd as f16 [b][s][d]
#pragma unroll
  for (int i = 0; i < 8; ++i) l[i] = 1.0f / l[i];
#pragma unroll
  for (int n = 0; n < 4; ++n)
#pragma unroll
    for (int i = 0; i < 8; ++i) {
      int r = i + 8 * hi;
      Ug[((size_t)b * S_ + q0 + r) * D_ + n * 16 + ln] = (_Float16)(acc[n][i] * l[i]);
    }
}

// ---------------------------------------------------------------------------
// Kernel 3: FFN  out = relu(upd@W1 + b1)@W2 + b2, all matmuls via WMMA.
// Dynamic LDS: W1T (32KB) + W2T (32KB) + per-wave bounce (8KB) = 73728 B.
// ---------------------------------------------------------------------------
__global__ __launch_bounds__(256) void enc_ffn_kernel(
    const _Float16* __restrict__ Ug, const float* __restrict__ w1,
    const float* __restrict__ b1, const float* __restrict__ w2,
    const float* __restrict__ b2, float* __restrict__ out) {
  extern __shared__ __align__(32) char smem[];
  _Float16* W1T = (_Float16*)smem;                 // [256][64]
  _Float16* W2T = (_Float16*)(smem + 32768);       // [64][256]
  _Float16* Pb  = (_Float16*)(smem + 65536);       // [8][16*32]

  const int tid  = threadIdx.x;
  const int lane = tid & 31;
  const int wave = tid >> 5;
  const int ln   = lane & 15;
  const int hi   = lane >> 4;

  for (int idx = tid; idx < 64 * 256; idx += 256) {
    int d = idx >> 8, n = idx & 255;
    W1T[n * 64 + d] = (_Float16)w1[idx];
  }
  for (int idx = tid; idx < 256 * 64; idx += 256) {
    int h = idx >> 6, n = idx & 63;
    W2T[n * 256 + h] = (_Float16)w2[idx];
  }
  __syncthreads();

  const int tokBase = blockIdx.x * 128 + wave * 16;

  // upd A-fragments
  const _Float16* urow = Ug + (size_t)(tokBase + ln) * D_;
  HV ua[2];
#pragma unroll
  for (int c = 0; c < 2; ++c) {
    int cb = 32 * c + hi * 16;
    ua[c].h[0] = *(const v8h*)(urow + cb);
    ua[c].h[1] = *(const v8h*)(urow + cb + 8);
  }

  v8f oacc[4];
#pragma unroll
  for (int n = 0; n < 4; ++n) oacc[n] = vzero8();
  _Float16* pb = Pb + wave * 512;

#pragma unroll 1
  for (int kc = 0; kc < 8; ++kc) {
    // two 16-col tiles of h = relu(upd@W1 + b1), cols [kc*32, kc*32+32)
#pragma unroll
    for (int t = 0; t < 2; ++t) {
      int nt = kc * 2 + t;
      v8f hacc = vzero8();
#pragma unroll
      for (int c = 0; c < 2; ++c) {
        v16h bf = *(const v16h*)(W1T + (nt * 16 + ln) * 64 + c * 32 + hi * 16);
        hacc = wmma_f16(ua[c].v, bf, hacc);
      }
      float bias = b1[nt * 16 + ln];
#pragma unroll
      for (int i = 0; i < 8; ++i) {
        float v = hacc[i] + bias;
        v = v > 0.0f ? v : 0.0f;
        int r = i + 8 * hi;
        pb[r * 32 + t * 16 + ln] = (_Float16)v;
      }
    }
    wait_ds0();
    HV ha;
    ha.h[0] = *(const v8h*)(pb + ln * 32 + hi * 16);
    ha.h[1] = *(const v8h*)(pb + ln * 32 + hi * 16 + 8);
    // out += h_chunk @ W2_chunk
#pragma unroll
    for (int n = 0; n < 4; ++n) {
      v16h bf = *(const v16h*)(W2T + (n * 16 + ln) * 256 + kc * 32 + hi * 16);
      oacc[n] = wmma_f16(ha.v, bf, oacc[n]);
    }
  }

#pragma unroll
  for (int n = 0; n < 4; ++n) {
    float bias = b2[n * 16 + ln];
#pragma unroll
    for (int i = 0; i < 8; ++i) {
      int r = i + 8 * hi;
      out[(size_t)(tokBase + r) * D_ + n * 16 + ln] = oacc[n][i] + bias;
    }
  }
}

extern "C" void kernel_launch(void* const* d_in, const int* in_sizes, int n_in,
                              void* d_out, int out_size, void* d_ws, size_t ws_size,
                              hipStream_t stream) {
  (void)in_sizes; (void)n_in; (void)out_size; (void)ws_size;
  const int*   x   = (const int*)d_in[0];
  const float* emb = (const float*)d_in[1];
  const float* wq  = (const float*)d_in[2];
  const float* bq  = (const float*)d_in[3];
  const float* wk  = (const float*)d_in[4];
  const float* bk  = (const float*)d_in[5];
  const float* wv  = (const float*)d_in[6];
  const float* bv  = (const float*)d_in[7];
  const float* w1  = (const float*)d_in[8];
  const float* b1  = (const float*)d_in[9];
  const float* w2  = (const float*)d_in[10];
  const float* b2  = (const float*)d_in[11];
  float* out = (float*)d_out;

  const size_t N = (size_t)B_ * S_ * D_;
  _Float16* Qg = (_Float16*)d_ws;
  _Float16* Kg = Qg + N;
  _Float16* Vg = Kg + N;
  _Float16* Ug = Vg + N;

  const int blocks = (B_ * S_) / 128;  // 512
  enc_qkv_kernel<<<dim3(blocks), dim3(256), 0, stream>>>(
      x, emb, wq, bq, wk, bk, wv, bv, Qg, Kg, Vg);
  enc_attn_kernel<<<dim3(blocks), dim3(256), 0, stream>>>(Qg, Kg, Vg, Ug);
  enc_ffn_kernel<<<dim3(blocks), dim3(256), 73728, stream>>>(
      Ug, w1, b1, w2, b2, out);
}
